// LuongAttention_21672404975617
// MI455X (gfx1250) — matrix operand order repaired and checked by reading, not compiled, not run on hardware
//
#include <hip/hip_runtime.h>
#include <hip/hip_bf16.h>

#define B_  16
#define SE_ 2048
#define SD_ 1024
#define H_  1024

typedef unsigned short ushort_t;
typedef __attribute__((ext_vector_type(16))) __bf16          bf16x16;
typedef __attribute__((ext_vector_type(8)))  unsigned short  ushort8;
typedef __attribute__((ext_vector_type(8)))  float           floatx8;
typedef __attribute__((ext_vector_type(4)))  float           floatx4;

__device__ __forceinline__ unsigned short bf16_bits(float f) {
  __bf16 h = (__bf16)f;                       // RNE; should lower to v_cvt_*bf16_f32
  return __builtin_bit_cast(unsigned short, h);
}

__device__ __forceinline__ floatx8 wmma_bf16(bf16x16 a, bf16x16 b, floatx8 c) {
  return __builtin_amdgcn_wmma_f32_16x16x32_bf16(false, a, false, b, (short)0, c,
                                                 false, false);
}

// A fragment (16x32, lane m = row, 16 consecutive K at kg*16) from f32 row-major.
__device__ __forceinline__ bf16x16 load_a_f32(const float* __restrict__ A, size_t row0,
                                              int k0, int ld, int lane) {
  const int m = lane & 15, kg = (lane >> 4) & 1;
  const float* p = A + (row0 + (size_t)m) * (size_t)ld + (size_t)(k0 + kg * 16);
  floatx4 f0 = *(const floatx4*)(p);
  floatx4 f1 = *(const floatx4*)(p + 4);
  floatx4 f2 = *(const floatx4*)(p + 8);
  floatx4 f3 = *(const floatx4*)(p + 12);
  bf16x16 r;
#pragma unroll
  for (int i = 0; i < 4; ++i) {
    r[i]      = (__bf16)f0[i];
    r[4 + i]  = (__bf16)f1[i];
    r[8 + i]  = (__bf16)f2[i];
    r[12 + i] = (__bf16)f3[i];
  }
  return r;
}

// B fragment (32x16, lane n = column, 16 consecutive K) from bf16 [N][K] row-major.
__device__ __forceinline__ bf16x16 load_b_bf16t(const ushort_t* __restrict__ Bt,
                                                size_t n0, int k0, int ld, int lane) {
  const int n = lane & 15, kg = (lane >> 4) & 1;
  const ushort_t* p = Bt + (n0 + (size_t)n) * (size_t)ld + (size_t)(k0 + kg * 16);
  union { bf16x16 bf; struct { ushort8 lo, hi; } h; } u;
  u.h.lo = *(const ushort8*)(p);
  u.h.hi = *(const ushort8*)(p + 8);
  return u.bf;
}

// ---------------- W transpose: Wt[k][h] = bf16(W[h][k]) ----------------
__global__ __launch_bounds__(256) void k_transpose_w(const float* __restrict__ W,
                                                     ushort_t* __restrict__ Wt) {
  __shared__ float tile[32][33];
  const int tx = threadIdx.x;
  const int x  = blockIdx.x * 32 + tx;        // k index in W
  const int y0 = blockIdx.y * 32;             // h base
  for (int i = threadIdx.y; i < 32; i += 8)
    tile[i][tx] = W[(size_t)(y0 + i) * H_ + x];
  __syncthreads();
  const int xo  = blockIdx.y * 32 + tx;       // h as fast dim of Wt
  const int ko0 = blockIdx.x * 32;
  for (int i = threadIdx.y; i < 32; i += 8)
    Wt[(size_t)(ko0 + i) * H_ + xo] = bf16_bits(tile[tx][i]);
}

// Common 256x128 block / 64x64 wave GEMM body: C(f32 src A) = A @ Bt^T
// A: f32 row-major [M][K] (ld=ldk), Bt: bf16 row-major [N][K] (ld=ldk).
// Epilogue handled by callers.

// ---------------- z = enc @ W + b  (bf16 out to workspace) ----------------
__global__ __launch_bounds__(256) void k_gemm_z(const float* __restrict__ enc,
                                                const ushort_t* __restrict__ Wt,
                                                const float* __restrict__ bias,
                                                ushort_t* __restrict__ z) {
  const int lane = threadIdx.x & 31;
  const int wave = threadIdx.x >> 5;
  const int wm = wave >> 1, wn = wave & 1;
  const size_t m0 = (size_t)blockIdx.x * 256 + wm * 64;
  const int    n0 = blockIdx.y * 128 + wn * 64;
  floatx8 acc[4][4];
#pragma unroll
  for (int i = 0; i < 4; ++i)
#pragma unroll
    for (int j = 0; j < 4; ++j)
#pragma unroll
      for (int v = 0; v < 8; ++v) acc[i][j][v] = 0.f;

  for (int k0 = 0; k0 < H_; k0 += 32) {
    bf16x16 a[4];
#pragma unroll
    for (int i = 0; i < 4; ++i) a[i] = load_a_f32(enc, m0 + i * 16, k0, H_, lane);
#pragma unroll
    for (int j = 0; j < 4; ++j) {
      bf16x16 b = load_b_bf16t(Wt, (size_t)(n0 + j * 16), k0, H_, lane);
#pragma unroll
      for (int i = 0; i < 4; ++i) acc[i][j] = wmma_bf16(a[i], b, acc[i][j]);
    }
  }
  const int cl = lane & 15, rg = lane >> 4;
#pragma unroll
  for (int j = 0; j < 4; ++j) {
    const int col = n0 + j * 16 + cl;
    const float bv = bias[col];
#pragma unroll
    for (int i = 0; i < 4; ++i) {
      const size_t rb = m0 + i * 16 + rg * 8;
#pragma unroll
      for (int v = 0; v < 8; ++v)
        z[(rb + v) * (size_t)H_ + col] = bf16_bits(acc[i][j][v] + bv);
    }
  }
}

// ---------------- scores = dec @ z^T  (f32 out, pre-softmax) ----------------
__global__ __launch_bounds__(256) void k_gemm_scores(const float* __restrict__ dec,
                                                     const ushort_t* __restrict__ zbf,
                                                     float* __restrict__ sc) {
  const int lane = threadIdx.x & 31;
  const int wave = threadIdx.x >> 5;
  const int wm = wave >> 1, wn = wave & 1;
  const size_t bz = blockIdx.z;
  const float*    A  = dec + bz * (size_t)SD_ * H_;
  const ushort_t* Bt = zbf + bz * (size_t)SE_ * H_;
  float*          C  = sc  + bz * (size_t)SD_ * SE_;
  const size_t m0 = (size_t)blockIdx.x * 256 + wm * 64;
  const int    n0 = blockIdx.y * 128 + wn * 64;
  floatx8 acc[4][4];
#pragma unroll
  for (int i = 0; i < 4; ++i)
#pragma unroll
    for (int j = 0; j < 4; ++j)
#pragma unroll
      for (int v = 0; v < 8; ++v) acc[i][j][v] = 0.f;

  for (int k0 = 0; k0 < H_; k0 += 32) {
    bf16x16 a[4];
#pragma unroll
    for (int i = 0; i < 4; ++i) a[i] = load_a_f32(A, m0 + i * 16, k0, H_, lane);
#pragma unroll
    for (int j = 0; j < 4; ++j) {
      bf16x16 b = load_b_bf16t(Bt, (size_t)(n0 + j * 16), k0, H_, lane);
#pragma unroll
      for (int i = 0; i < 4; ++i) acc[i][j] = wmma_bf16(a[i], b, acc[i][j]);
    }
  }
  const int cl = lane & 15, rg = lane >> 4;
#pragma unroll
  for (int j = 0; j < 4; ++j) {
    const int col = n0 + j * 16 + cl;
#pragma unroll
    for (int i = 0; i < 4; ++i) {
      const size_t rb = m0 + i * 16 + rg * 8;
#pragma unroll
      for (int v = 0; v < 8; ++v)
        C[(rb + v) * (size_t)SE_ + col] = acc[i][j][v];
    }
  }
}

// ---------------- row softmax over S_enc (in-place on weights) ----------------
__global__ __launch_bounds__(256) void k_softmax(float* __restrict__ w) {
  __shared__ float red[8];
  const size_t row = blockIdx.x;
  float* p = w + row * (size_t)SE_;
  const int t = threadIdx.x, lane = t & 31, wave = t >> 5;
  float v[8];
  float m = -3.4e38f;
#pragma unroll
  for (int i = 0; i < 8; ++i) { v[i] = p[t + i * 256]; m = fmaxf(m, v[i]); }
#pragma unroll
  for (int off = 16; off >= 1; off >>= 1) m = fmaxf(m, __shfl_xor(m, off, 32));
  if (lane == 0) red[wave] = m;
  __syncthreads();
  m = red[0];
#pragma unroll
  for (int i = 1; i < 8; ++i) m = fmaxf(m, red[i]);
  __syncthreads();
  float s = 0.f;
#pragma unroll
  for (int i = 0; i < 8; ++i) { v[i] = __expf(v[i] - m); s += v[i]; }
#pragma unroll
  for (int off = 16; off >= 1; off >>= 1) s += __shfl_xor(s, off, 32);
  if (lane == 0) red[wave] = s;
  __syncthreads();
  s = red[0];
#pragma unroll
  for (int i = 1; i < 8; ++i) s += red[i];
  const float inv = 1.f / s;
#pragma unroll
  for (int i = 0; i < 8; ++i) p[t + i * 256] = v[i] * inv;
}

// ---------------- context = weights @ enc (enc tile transposed via LDS) ------
__global__ __launch_bounds__(256) void k_gemm_ctx(const float* __restrict__ wgt,
                                                  const float* __restrict__ enc,
                                                  float* __restrict__ ctx) {
  __shared__ __align__(16) ushort_t Bs[128 * 40];   // [n_local][k_local], 80B rows
  const int lane = threadIdx.x & 31;
  const int wave = threadIdx.x >> 5;
  const int wm = wave >> 1, wn = wave & 1;
  const size_t bz = blockIdx.z;
  const float* A = wgt + bz * (size_t)SD_ * SE_;
  const float* E = enc + bz * (size_t)SE_ * H_;
  float*       C = ctx + bz * (size_t)SD_ * H_;
  const int bn = blockIdx.y * 128;
  const size_t m0 = (size_t)blockIdx.x * 256 + wm * 64;
  const int n0l = wn * 64;
  floatx8 acc[4][4];
#pragma unroll
  for (int i = 0; i < 4; ++i)
#pragma unroll
    for (int j = 0; j < 4; ++j)
#pragma unroll
      for (int v = 0; v < 8; ++v) acc[i][j][v] = 0.f;

  for (int k0 = 0; k0 < SE_; k0 += 32) {
    __syncthreads();
    // stage enc[k0..k0+31][bn..bn+127] transposed into Bs as bf16
#pragma unroll
    for (int it = 0; it < 4; ++it) {
      const int idx = threadIdx.x + it * 256;   // 1024 float4 tiles
      const int er  = idx >> 5;                 // e-local 0..31
      const int c4  = idx & 31;                 // h-local/4
      floatx4 f = *(const floatx4*)(E + (size_t)(k0 + er) * H_ + bn + c4 * 4);
#pragma unroll
      for (int q = 0; q < 4; ++q)
        Bs[(c4 * 4 + q) * 40 + er] = bf16_bits(f[q]);
    }
    __syncthreads();
    bf16x16 a[4];
#pragma unroll
    for (int i = 0; i < 4; ++i) a[i] = load_a_f32(A, m0 + i * 16, k0, SE_, lane);
    const int nl = lane & 15, kg = (lane >> 4) & 1;
#pragma unroll
    for (int j = 0; j < 4; ++j) {
      const ushort_t* p = &Bs[(n0l + j * 16 + nl) * 40 + kg * 16];
      union { bf16x16 bf; struct { ushort8 lo, hi; } h; } u;
      u.h.lo = *(const ushort8*)(p);
      u.h.hi = *(const ushort8*)(p + 8);
#pragma unroll
      for (int i = 0; i < 4; ++i) acc[i][j] = wmma_bf16(a[i], u.bf, acc[i][j]);
    }
  }
  const int cl = lane & 15, rg = lane >> 4;
#pragma unroll
  for (int j = 0; j < 4; ++j) {
    const int col = bn + n0l + j * 16 + cl;
#pragma unroll
    for (int i = 0; i < 4; ++i) {
      const size_t rb = m0 + i * 16 + rg * 8;
#pragma unroll
      for (int v = 0; v < 8; ++v)
        C[(rb + v) * (size_t)H_ + col] = acc[i][j][v];
    }
  }
}

extern "C" void kernel_launch(void* const* d_in, const int* in_sizes, int n_in,
                              void* d_out, int out_size, void* d_ws, size_t ws_size,
                              hipStream_t stream) {
  (void)in_sizes; (void)n_in; (void)out_size; (void)ws_size;
  const float* enc  = (const float*)d_in[0];   // [16,2048,1024]
  const float* dec  = (const float*)d_in[1];   // [16,1024,1024]
  const float* W    = (const float*)d_in[2];   // [1024,1024]
  const float* bias = (const float*)d_in[3];   // [1024]
  float* weights = (float*)d_out;                              // [16,1024,2048]
  float* context = weights + (size_t)B_ * SD_ * SE_;           // [16,1024,1024]
  ushort_t* Wt  = (ushort_t*)d_ws;                             // 1024*1024 bf16 (2 MB)
  ushort_t* zbf = Wt + (size_t)H_ * H_;                        // 16*2048*1024 bf16 (64 MB)

  k_transpose_w<<<dim3(H_ / 32, H_ / 32), dim3(32, 8), 0, stream>>>(W, Wt);
  k_gemm_z<<<dim3((B_ * SE_) / 256, H_ / 128), 256, 0, stream>>>(enc, Wt, bias, zbf);
  k_gemm_scores<<<dim3(SD_ / 256, SE_ / 128, B_), 256, 0, stream>>>(dec, zbf, weights);
  k_softmax<<<dim3(B_ * SD_), 256, 0, stream>>>(weights);
  k_gemm_ctx<<<dim3(SD_ / 256, H_ / 128, B_), 256, 0, stream>>>(weights, enc, context);
}
